// _RBFKernel_65481071405263
// MI455X (gfx1250) — matrix operand order repaired and checked by reading, not compiled, not run on hardware
//
#include <hip/hip_runtime.h>
#include <hip/hip_bf16.h>

typedef __attribute__((ext_vector_type(16))) __bf16 v16bf;
typedef __attribute__((ext_vector_type(8)))  __bf16 v8bf;
typedef __attribute__((ext_vector_type(4)))  __bf16 v4bf;
typedef __attribute__((ext_vector_type(8)))  float  v8f;

#define NROWS 8192
#define DK    256
#define OUTLD 8192

#define LDSLD 40                         // padded row stride (bf16): 80B, 16B-aligned,
                                         // 20-bank stride -> conflict-free b128 reads
#define LDS_MAT   (128 * LDSLD)          // one matrix part (elements)
#define LDS_CHUNK (4 * LDS_MAT)          // Ahi,Alo,Bhi,Blo for one K-chunk
#define LDS_BYTES (2 * LDS_CHUNK * 2)    // double buffered, bf16 -> 81920 B

union FragU { v16bf v; v8bf h[2]; };

// ---------------------------------------------------------------------------
// Pass 1: row squared-norms for x and y into workspace.
// ---------------------------------------------------------------------------
__global__ void rbf_norms_kernel(const float* __restrict__ x,
                                 const float* __restrict__ y,
                                 float* __restrict__ xsq,
                                 float* __restrict__ ysq) {
    int idx = blockIdx.x * blockDim.x + threadIdx.x;   // 0 .. 2*NROWS-1
    if (idx >= 2 * NROWS) return;
    const float* src = (idx < NROWS) ? x : y;
    int row = (idx < NROWS) ? idx : idx - NROWS;
    const float4* p = reinterpret_cast<const float4*>(src + (long)row * DK);
    float s = 0.0f;
#pragma unroll 8
    for (int i = 0; i < DK / 4; ++i) {
        float4 v = p[i];
        s += v.x * v.x + v.y * v.y + v.z * v.z + v.w * v.w;
    }
    if (idx < NROWS) xsq[row] = s; else ysq[row] = s;
}

// ---------------------------------------------------------------------------
// Pass 1b: one-time f32 -> (bf16 hi, bf16 lo) split of x and y.
// ---------------------------------------------------------------------------
__global__ void rbf_convert_kernel(const float* __restrict__ x,
                                   const float* __restrict__ y,
                                   __bf16* __restrict__ xhi, __bf16* __restrict__ xlo,
                                   __bf16* __restrict__ yhi, __bf16* __restrict__ ylo) {
    const long total = (long)NROWS * DK / 4;           // float4s per matrix
    long idx = (long)blockIdx.x * blockDim.x + threadIdx.x;
    if (idx >= 2 * total) return;
    const float4* src;
    __bf16 *hi, *lo;
    long t;
    if (idx < total) { src = (const float4*)x; hi = xhi; lo = xlo; t = idx; }
    else             { src = (const float4*)y; hi = yhi; lo = ylo; t = idx - total; }
    float4 f = src[t];
    float v[4] = {f.x, f.y, f.z, f.w};
    v4bf h4, l4;
#pragma unroll
    for (int e = 0; e < 4; ++e) {
        __bf16 h = (__bf16)v[e];
        h4[e] = h;
        l4[e] = (__bf16)(v[e] - (float)h);
    }
    *reinterpret_cast<v4bf*>(hi + t * 4) = h4;
    *reinterpret_cast<v4bf*>(lo + t * 4) = l4;
}

// ---------------------------------------------------------------------------
// Async-stage one 32-K chunk (A 128 rows + B 128 rows, hi+lo) into LDS.
// 2048 x b128 transfers, 8 per thread, tracked by ASYNCcnt.
// ---------------------------------------------------------------------------
__device__ __forceinline__ void stage_chunk(const __bf16* __restrict__ Xhi,
                                            const __bf16* __restrict__ Xlo,
                                            const __bf16* __restrict__ Yhi,
                                            const __bf16* __restrict__ Ylo,
                                            __bf16* sbuf, int nBase, int mBase,
                                            int k0, int tid) {
#pragma unroll
    for (int q = 0; q < 8; ++q) {
        int id    = tid + 256 * q;       // 0..2047
        int rowid = id >> 2;             // 0..511
        int part  = id & 3;              // which 16B of the 64B row-chunk
        int sel   = rowid >> 7;          // 0:Ahi 1:Alo 2:Bhi 3:Blo
        int rr    = rowid & 127;
        const __bf16* gsrc =
            (sel == 0) ? Xhi + (long)(nBase + rr) * DK + k0 :
            (sel == 1) ? Xlo + (long)(nBase + rr) * DK + k0 :
            (sel == 2) ? Yhi + (long)(mBase + rr) * DK + k0 :
                         Ylo + (long)(mBase + rr) * DK + k0;
        gsrc += part * 8;
        __bf16* ldst = sbuf + sel * LDS_MAT + rr * LDSLD + part * 8;
        uint32_t laddr = (uint32_t)(size_t)ldst;     // low 32b = LDS offset
        uint64_t gaddr = (uint64_t)(size_t)gsrc;
        asm volatile("global_load_async_to_lds_b128 %0, %1, off"
                     :: "v"(laddr), "v"(gaddr) : "memory");
    }
}

// ---------------------------------------------------------------------------
// Pass 2: LDS-staged, double-buffered split-bf16 WMMA GEMM + RBF epilogue.
//   block: 256 threads = 8 waves (4 n x 2 m); wave tile 32x64; block 128x128.
// ---------------------------------------------------------------------------
__global__ void __launch_bounds__(256, 2)
rbf_wmma_kernel(const __bf16* __restrict__ Xhi, const __bf16* __restrict__ Xlo,
                const __bf16* __restrict__ Yhi, const __bf16* __restrict__ Ylo,
                const float* __restrict__ gptr,
                const float* __restrict__ xsq, const float* __restrict__ ysq,
                float* __restrict__ out) {
    extern __shared__ __bf16 smem[];     // 2 * LDS_CHUNK elements (80 KB)

    const int tid  = threadIdx.x;
    const int lane = tid & 31;
    const int wid  = tid >> 5;
    const int wn   = wid >> 1;        // 0..3
    const int wm   = wid & 1;         // 0..1
    const int r    = lane & 15;
    const int half = lane >> 4;

    const int nBase = blockIdx.y * 128;
    const int mBase = blockIdx.x * 128;
    const int n0 = nBase + wn * 32;
    const int m0 = mBase + wm * 64;

    v8f acc[2][4];
#pragma unroll
    for (int i = 0; i < 2; ++i)
#pragma unroll
        for (int j = 0; j < 4; ++j)
            acc[i][j] = {};

    stage_chunk(Xhi, Xlo, Yhi, Ylo, smem, nBase, mBase, 0, tid);

    for (int kc = 0; kc < DK / 32; ++kc) {
        asm volatile("s_wait_asynccnt 0" ::: "memory");
        __syncthreads();

        __bf16* cur = smem + (kc & 1) * LDS_CHUNK;
        if (kc + 1 < DK / 32)
            stage_chunk(Xhi, Xlo, Yhi, Ylo, smem + ((kc + 1) & 1) * LDS_CHUNK,
                        nBase, mBase, (kc + 1) * 32, tid);

        const __bf16* sAhi = cur;
        const __bf16* sAlo = cur + LDS_MAT;
        const __bf16* sBhi = cur + 2 * LDS_MAT;
        const __bf16* sBlo = cur + 3 * LDS_MAT;

        // A fragments (16x32 lane layout): half=0 -> K{0..7,16..23}, half=1 -> K{8..15,24..31}
        FragU ah[2], al[2];
#pragma unroll
        for (int i = 0; i < 2; ++i) {
            const int rowOff = (wn * 32 + i * 16 + r) * LDSLD;
            ah[i].h[0] = *reinterpret_cast<const v8bf*>(sAhi + rowOff + half * 8);
            ah[i].h[1] = *reinterpret_cast<const v8bf*>(sAhi + rowOff + 16 + half * 8);
            al[i].h[0] = *reinterpret_cast<const v8bf*>(sAlo + rowOff + half * 8);
            al[i].h[1] = *reinterpret_cast<const v8bf*>(sAlo + rowOff + 16 + half * 8);
        }
        // B fragments (32x16 lane layout): half=0 -> K{0..7,16..23}, half=1 -> K{8..15,24..31}
        FragU bh[4], bl[4];
#pragma unroll
        for (int j = 0; j < 4; ++j) {
            const int rowOff = (wm * 64 + j * 16 + r) * LDSLD;
            bh[j].h[0] = *reinterpret_cast<const v8bf*>(sBhi + rowOff + half * 8);
            bh[j].h[1] = *reinterpret_cast<const v8bf*>(sBhi + rowOff + 16 + half * 8);
            bl[j].h[0] = *reinterpret_cast<const v8bf*>(sBlo + rowOff + half * 8);
            bl[j].h[1] = *reinterpret_cast<const v8bf*>(sBlo + rowOff + 16 + half * 8);
        }

        // dot = hi*hi + hi*lo + lo*hi  (lo*lo ~ 2^-16, dropped)
#pragma unroll
        for (int i = 0; i < 2; ++i) {
#pragma unroll
            for (int j = 0; j < 4; ++j) {
                acc[i][j] = __builtin_amdgcn_wmma_f32_16x16x32_bf16(
                    false, ah[i].v, false, bh[j].v, (short)0, acc[i][j], false, false);
                acc[i][j] = __builtin_amdgcn_wmma_f32_16x16x32_bf16(
                    false, ah[i].v, false, bl[j].v, (short)0, acc[i][j], false, false);
                acc[i][j] = __builtin_amdgcn_wmma_f32_16x16x32_bf16(
                    false, al[i].v, false, bh[j].v, (short)0, acc[i][j], false, false);
            }
        }
    }

    // RBF epilogue. C/D layout: lanes 0-15: N=lane, VGPR rr -> M=rr;
    //                           lanes 16-31: N=lane-16, VGPR rr -> M=rr+8.
    const float gamma = gptr[0];
#pragma unroll
    for (int i = 0; i < 2; ++i) {
#pragma unroll
        for (int j = 0; j < 4; ++j) {
            const int gm = m0 + j * 16 + r;
            const float ys = ysq[gm];
#pragma unroll
            for (int rr = 0; rr < 8; ++rr) {
                const int gn = n0 + i * 16 + rr + 8 * half;
                float d = xsq[gn] + ys - 2.0f * acc[i][j][rr];
                d = fmaxf(d, 0.0f);
                __builtin_nontemporal_store(__expf(-gamma * d),
                                            out + (long)gn * OUTLD + gm);
            }
        }
    }
}

// ---------------------------------------------------------------------------
extern "C" void kernel_launch(void* const* d_in, const int* in_sizes, int n_in,
                              void* d_out, int out_size, void* d_ws, size_t ws_size,
                              hipStream_t stream) {
    const float* x     = (const float*)d_in[0];
    const float* y     = (const float*)d_in[1];
    const float* gamma = (const float*)d_in[2];
    float* out = (float*)d_out;

    const size_t matElems = (size_t)NROWS * DK;        // 2M elements
    __bf16* xhi = (__bf16*)d_ws;
    __bf16* xlo = xhi + matElems;
    __bf16* yhi = xlo + matElems;
    __bf16* ylo = yhi + matElems;
    float*  xsq = (float*)(ylo + matElems);
    float*  ysq = xsq + NROWS;
    // total ws use: 16 MB + 64 KB

    // Allow 80 KB dynamic LDS (WGP has 320 KB). Deterministic, capture-safe.
    hipFuncSetAttribute((const void*)rbf_wmma_kernel,
                        hipFuncAttributeMaxDynamicSharedMemorySize, LDS_BYTES);

    // Pass 1: norms
    {
        int threads = 2 * NROWS;
        dim3 blk(256), grd((threads + 255) / 256);
        rbf_norms_kernel<<<grd, blk, 0, stream>>>(x, y, xsq, ysq);
    }
    // Pass 1b: one-time bf16 hi/lo split
    {
        long threads = 2L * matElems / 4;
        dim3 blk(256), grd((unsigned)((threads + 255) / 256));
        rbf_convert_kernel<<<grd, blk, 0, stream>>>(x, y, xhi, xlo, yhi, ylo);
    }
    // Pass 2: async-staged WMMA cross-GEMM + epilogue
    {
        dim3 blk(256);
        dim3 grd(NROWS / 128, NROWS / 128);
        rbf_wmma_kernel<<<grd, blk, LDS_BYTES, stream>>>(xhi, xlo, yhi, ylo,
                                                         gamma, xsq, ysq, out);
    }
}
